// GATPortfolio_39444979646659
// MI455X (gfx1250) — matrix operand order repaired and checked by reading, not compiled, not run on hardware
//
#include <hip/hip_runtime.h>
#include <math.h>

// ---------------------------------------------------------------------------
// Types for CDNA5 WMMA (wave32, 16x16x32 bf16 -> f32 accumulate)
// ---------------------------------------------------------------------------
typedef __attribute__((ext_vector_type(16))) __bf16 v16bf;
typedef __attribute__((ext_vector_type(8)))  float  v8f;

__device__ __forceinline__ __bf16 f2bf(float f) { return (__bf16)f; }  // native v_cvt, RNE

// ordered-int encoding so atomicMax(int) == float max
__device__ __forceinline__ int f2o(float f) {
  int i = __float_as_int(f);
  return i >= 0 ? i : (int)(i ^ 0x7FFFFFFF);
}
__device__ __forceinline__ float o2f(int i) {
  return __int_as_float(i >= 0 ? i : (int)(i ^ 0x7FFFFFFF));
}

#define ACT_NONE 0
#define ACT_RELU 1

// ---------------------------------------------------------------------------
// WMMA GEMM: Y[M,Nout] = act( (X[M,K]+abias[K]) @ W[Nout,K]^T + bias[Nout] + res[M,Nout] )
// K compile-time (64/128/256) -> fully unrolled WMMA chain.
// blockDim = (32,4): 4 waves, one 16x16 output tile each (same 16 cols, 4 M-tiles).
// The 16xK weight tile is staged once per block into LDS with
// global_load_async_to_lds_b128 (ASYNCcnt) and consumed via ds_load.
// ---------------------------------------------------------------------------
template <int K>
__global__ __launch_bounds__(128) void wmma_gemm(
    const float* __restrict__ X, const float* __restrict__ W,
    const float* __restrict__ bias, const float* __restrict__ abias,
    const float* __restrict__ res, float* __restrict__ Y,
    int M, int Nout, int act)
{
  __shared__ float lw[16 * K];                 // weight tile: 16 rows x K
  int tn   = blockIdx.x;
  int tm   = blockIdx.y * 4 + threadIdx.y;
  int lane = threadIdx.x;                      // 0..31
  int grp  = lane >> 4;                        // half-wave 0/1
  int l16  = lane & 15;
  bool active = (tm * 16 < M);                 // tail waves still do staging+barrier

  // ---- async stage W tile (16 x K floats) into LDS, cooperatively ----
  {
    int tid = threadIdx.y * 32 + threadIdx.x;  // 0..127
    const int CH = (16 * K) / 4;               // float4 chunks
#pragma unroll
    for (int c0 = 0; c0 < CH; c0 += 128) {
      int c   = c0 + tid;
      int row = c / (K / 4);
      int kc  = c % (K / 4);
      int wr  = tn * 16 + row; if (wr >= Nout) wr = Nout - 1;
      unsigned long long gp =
          (unsigned long long)(const void*)(W + (size_t)wr * K + (size_t)kc * 4);
      unsigned lds_off = (unsigned)(unsigned long long)(const void*)&lw[row * K + kc * 4];
      asm volatile("global_load_async_to_lds_b128 %0, %1, off"
                   :: "v"(lds_off), "v"(gp) : "memory");
    }
    asm volatile("s_wait_asynccnt 0" ::: "memory");
    __syncthreads();
  }

  if (active) {
    int m_a = tm * 16 + l16; if (m_a >= M) m_a = M - 1;
    const float* __restrict__ arow = X + (size_t)m_a * K;
    const float* __restrict__ lrow = lw + l16 * K;   // this lane's output column

    v8f acc = {};
#pragma unroll
    for (int k0 = 0; k0 < K; k0 += 32) {
      // A fragment (16x32 bf16, ISA 7.12.2): halves 0..7 -> K=ka..ka+7,
      // halves 8..15 -> K=ka+16..ka+23, ka = k0 + grp*8
      int ka = k0 + grp * 8;
      // B fragment (32x16 bf16): lane holds col l16; same K order as A
      int kb = k0 + grp * 8;
      v16bf a, b;
      if (abias) {
#pragma unroll
        for (int j = 0; j < 8; j++) {
          a[j]     = f2bf(arow[ka + j]      + abias[ka + j]);
          a[8 + j] = f2bf(arow[ka + 16 + j] + abias[ka + 16 + j]);
        }
      } else {
#pragma unroll
        for (int j = 0; j < 8; j++) {
          a[j]     = f2bf(arow[ka + j]);
          a[8 + j] = f2bf(arow[ka + 16 + j]);
        }
      }
#pragma unroll
      for (int j = 0; j < 16; j++) b[j] = f2bf(lrow[kb + (j < 8 ? j : j + 8)]);

      if (k0 + 32 < K) __builtin_prefetch(arow + k0 + 32, 0, 0);  // global_prefetch_b8
      acc = __builtin_amdgcn_wmma_f32_16x16x32_bf16(
          false, a, false, b, (short)0, acc, false, false);
    }

    int ncol = tn * 16 + l16;
    float bv = bias ? bias[ncol < Nout ? ncol : 0] : 0.f;
#pragma unroll
    for (int v = 0; v < 8; v++) {
      int mrow = tm * 16 + grp * 8 + v;
      if (mrow >= M || ncol >= Nout) continue;
      float val = acc[v] + bv;
      if (res) val += res[(size_t)mrow * Nout + ncol];
      if (act == ACT_RELU) val = fmaxf(val, 0.f);
      Y[(size_t)mrow * Nout + ncol] = val;
    }
  }
}

// ---------------------------------------------------------------------------
// Small utility / graph kernels
// ---------------------------------------------------------------------------
__global__ void fill_u32(unsigned* __restrict__ p, unsigned v, int n) {
  int i = blockIdx.x * blockDim.x + threadIdx.x;
  if (i < n) p[i] = v;
}

__global__ void loop_stats(const int* __restrict__ src, const int* __restrict__ dst,
                           const float* __restrict__ ea, float* __restrict__ cnt,
                           float* __restrict__ sm, int E) {
  int e = blockIdx.x * blockDim.x + threadIdx.x;
  if (e >= E) return;
  int s = src[e], d = dst[e];
  if (s != d) {
    atomicAdd(&cnt[d], 1.f);
    atomicAdd(&sm[d], ea[e]);
  }
}
__global__ void loop_div(const float* __restrict__ cnt, float* __restrict__ loop_ea, int n) {
  int i = blockIdx.x * blockDim.x + threadIdx.x;
  if (i < n) loop_ea[i] = loop_ea[i] / fmaxf(cnt[i], 1.f);
}

// GATv2 logits per (edge, head) + ordered-int atomic segment max
__global__ void logits_kernel(const float* __restrict__ xl, const float* __restrict__ xr,
                              const int* __restrict__ src, const int* __restrict__ dst,
                              const float* __restrict__ ea, const float* __restrict__ loop_ea,
                              const float* __restrict__ We, const float* __restrict__ att,
                              float* __restrict__ logits, int* __restrict__ mxo,
                              int E, int n) {
  int t = blockIdx.x * blockDim.x + threadIdx.x;
  int tot = (E + n) * 4;
  if (t >= tot) return;
  int e = t >> 2, h = t & 3;
  int s, d; float eav; bool valid;
  if (e < E) { s = src[e]; d = dst[e]; eav = ea[e]; valid = (s != d); }
  else       { s = d = e - E; eav = loop_ea[s]; valid = true; }
  const float* pl = xl + (size_t)s * 256 + h * 64;
  const float* pr = xr + (size_t)d * 256 + h * 64;
  const float* pw = We + h * 64;
  const float* pa = att + h * 64;
  float acc = 0.f;
  for (int c = 0; c < 64; c++) {
    float v = pl[c] + pr[c] + eav * pw[c];
    v = v > 0.f ? v : 0.2f * v;            // leaky_relu 0.2
    acc += v * pa[c];
  }
  logits[t] = valid ? acc : -__builtin_inff();
  if (valid) atomicMax(&mxo[d * 4 + h], f2o(acc));
}

__global__ void denom_kernel(const float* __restrict__ logits, const int* __restrict__ dst,
                             const int* __restrict__ mxo, float* __restrict__ denom,
                             int E, int n) {
  int t = blockIdx.x * blockDim.x + threadIdx.x;
  int tot = (E + n) * 4;
  if (t >= tot) return;
  int e = t >> 2, h = t & 3;
  int d = (e < E) ? dst[e] : (e - E);
  float ex = expf(logits[t] - o2f(mxo[d * 4 + h]));
  if (ex > 0.f) atomicAdd(&denom[d * 4 + h], ex);
}

__global__ void scatter_kernel(const float* __restrict__ xl, const float* __restrict__ logits,
                               const int* __restrict__ mxo, const float* __restrict__ denom,
                               const int* __restrict__ src, const int* __restrict__ dst,
                               float* __restrict__ agg, int E, int n) {
  int t = blockIdx.x * blockDim.x + threadIdx.x;
  int tot = (E + n) * 4;
  if (t >= tot) return;
  int e = t >> 2, h = t & 3;
  int s, d;
  if (e < E) { s = src[e]; d = dst[e]; } else { s = d = e - E; }
  float a = expf(logits[t] - o2f(mxo[d * 4 + h])) / denom[d * 4 + h];
  if (a == 0.f) return;
  const float* xs = xl + (size_t)s * 256 + h * 64;
  float* o = agg + (size_t)d * 256 + h * 64;
  for (int c = 0; c < 64; c++) atomicAdd(&o[c], xs[c] * a);
}

// ELU + LayerNorm over 64 channels, in place. blockDim=64, one node per block.
__global__ void elu_ln(float* __restrict__ h, const float* __restrict__ g,
                       const float* __restrict__ be, int n) {
  __shared__ float red[64];
  __shared__ float s_mu, s_var;
  int node = blockIdx.x, j = threadIdx.x;
  if (node >= n) return;
  float* row = h + (size_t)node * 64;
  float v = row[j];
  v = v > 0.f ? v : expm1f(v);
  red[j] = v; __syncthreads();
  for (int s = 32; s > 0; s >>= 1) { if (j < s) red[j] += red[j + s]; __syncthreads(); }
  if (j == 0) s_mu = red[0] * (1.f / 64.f);
  __syncthreads();
  float dv = v - s_mu;
  red[j] = dv * dv; __syncthreads();
  for (int s = 32; s > 0; s >>= 1) { if (j < s) red[j] += red[j + s]; __syncthreads(); }
  if (j == 0) s_var = red[0] * (1.f / 64.f);
  __syncthreads();
  row[j] = dv * rsqrtf(s_var + 1e-5f) * g[j] + be[j];
}

// GRUCell (torch gate order r,z,n); biases already folded into gi/gh GEMMs
__global__ void gru_kernel(const float* __restrict__ gi, const float* __restrict__ gh,
                           const float* __restrict__ pm, float* __restrict__ mnew, int n) {
  int t = blockIdx.x * blockDim.x + threadIdx.x;
  if (t >= n * 64) return;
  int node = t >> 6, j = t & 63;
  const float* i  = gi + (size_t)node * 192;
  const float* hh = gh + (size_t)node * 192;
  float r  = 1.f / (1.f + expf(-(i[j]       + hh[j])));
  float z  = 1.f / (1.f + expf(-(i[64 + j]  + hh[64 + j])));
  float ng = tanhf(i[128 + j] + r * hh[128 + j]);
  mnew[t] = (1.f - z) * ng + z * pm[t];
}

__global__ void cat_kernel(const float* __restrict__ h1, const float* __restrict__ mnew,
                           float* __restrict__ cat, int n) {
  int t = blockIdx.x * blockDim.x + threadIdx.x;
  if (t >= n * 128) return;
  int node = t >> 7, j = t & 127;
  cat[t] = (j < 64) ? h1[(size_t)node * 64 + j] : mnew[(size_t)node * 64 + (j - 64)];
}

__global__ void scores_kernel(const float* __restrict__ hr, const float* __restrict__ w,
                              const float* __restrict__ b, float* __restrict__ out, int n) {
  int i = blockIdx.x * blockDim.x + threadIdx.x;
  if (i >= n) return;
  const float* row = hr + (size_t)i * 64;
  float acc = b[0];
  for (int j = 0; j < 64; j++) acc += row[j] * w[j];
  out[i] = acc;
}

// Masked sparsemax via bisection on tau: sum(max(z - tau, 0)) == 1,
// tau in [zmax-1, zmax]. Single block of 1024 threads; scores stay L2-resident.
__global__ __launch_bounds__(1024) void sparsemax_kernel(
    const float* __restrict__ scores, const unsigned char* __restrict__ mask,
    float* __restrict__ wts, int n) {
  __shared__ float red[1024];
  __shared__ float s_lo, s_hi, s_inv;
  const float VNEG = -8.5070592e37f;   // finfo(float32).min / 4
  int tid = threadIdx.x;

  float mx = -__builtin_inff();
  for (int i = tid; i < n; i += 1024) {
    float z = mask[i] ? scores[i] : VNEG;
    mx = fmaxf(mx, z);
  }
  red[tid] = mx; __syncthreads();
  for (int s = 512; s > 0; s >>= 1) { if (tid < s) red[tid] = fmaxf(red[tid], red[tid + s]); __syncthreads(); }
  if (tid == 0) { s_hi = red[0]; s_lo = red[0] - 1.f; }
  __syncthreads();

  for (int it = 0; it < 60; it++) {
    float tau = 0.5f * (s_lo + s_hi);
    float sum = 0.f;
    for (int i = tid; i < n; i += 1024) {
      float z = mask[i] ? scores[i] : VNEG;
      sum += fmaxf(z - tau, 0.f);
    }
    red[tid] = sum; __syncthreads();
    for (int s = 512; s > 0; s >>= 1) { if (tid < s) red[tid] += red[tid + s]; __syncthreads(); }
    if (tid == 0) { if (red[0] >= 1.f) s_lo = tau; else s_hi = tau; }
    __syncthreads();
  }
  float tau = 0.5f * (s_lo + s_hi);

  float sum = 0.f;
  for (int i = tid; i < n; i += 1024) {
    float z = mask[i] ? scores[i] : VNEG;
    sum += fmaxf(z - tau, 0.f);
  }
  red[tid] = sum; __syncthreads();
  for (int s = 512; s > 0; s >>= 1) { if (tid < s) red[tid] += red[tid + s]; __syncthreads(); }
  if (tid == 0) s_inv = 1.f / fmaxf(red[0], 1e-12f);
  __syncthreads();
  float inv = s_inv;
  for (int i = tid; i < n; i += 1024) {
    float z = mask[i] ? scores[i] : VNEG;
    wts[i] = mask[i] ? fmaxf(z - tau, 0.f) * inv : 0.f;
  }
}

// ---------------------------------------------------------------------------
// Host-side orchestration
// ---------------------------------------------------------------------------
static inline int cdiv(int a, int b) { return (a + b - 1) / b; }

template <int K>
static void gemm_launch(hipStream_t st, const float* X, const float* W,
                        const float* bias, const float* abias, const float* res,
                        float* Y, int M, int Nout, int act) {
  dim3 grid(Nout / 16, cdiv(cdiv(M, 16), 4));
  dim3 blk(32, 4);
  wmma_gemm<K><<<grid, blk, 0, st>>>(X, W, bias, abias, res, Y, M, Nout, act);
}

extern "C" void kernel_launch(void* const* d_in, const int* in_sizes, int n_in,
                              void* d_out, int out_size, void* d_ws, size_t ws_size,
                              hipStream_t stream) {
  const int IN_DIM = 128, HID = 64, HC = 256, MEM = 64;
  const float* x    = (const float*)d_in[0];
  const int*   ei   = (const int*)d_in[1];
  const unsigned char* mask = (const unsigned char*)d_in[2];
  const float* ea   = (const float*)d_in[3];
  const float* pmem = (const float*)d_in[4];
  int N = in_sizes[0] / IN_DIM;
  int E = in_sizes[1] / 2;
  const int* src = ei;
  const int* dst = ei + E;

  const float *Wl[2], *bl[2], *Wr[2], *br[2], *Wee[2], *att[2], *bc[2], *Wp[2], *bp[2], *g[2], *be[2];
  for (int l = 0; l < 2; l++) {
    int b = 5 + l * 11;
    Wl[l]  = (const float*)d_in[b + 0];  bl[l] = (const float*)d_in[b + 1];
    Wr[l]  = (const float*)d_in[b + 2];  br[l] = (const float*)d_in[b + 3];
    Wee[l] = (const float*)d_in[b + 4];  att[l] = (const float*)d_in[b + 5];
    bc[l]  = (const float*)d_in[b + 6];  Wp[l] = (const float*)d_in[b + 7];
    bp[l]  = (const float*)d_in[b + 8];  g[l]  = (const float*)d_in[b + 9];
    be[l]  = (const float*)d_in[b + 10];
  }
  const float* gru_wih = (const float*)d_in[27];
  const float* gru_whh = (const float*)d_in[28];
  const float* gru_bih = (const float*)d_in[29];
  const float* gru_bhh = (const float*)d_in[30];
  const float* Wf  = (const float*)d_in[31];
  const float* bfb = (const float*)d_in[32];
  const float* r1w = (const float*)d_in[33];
  const float* r1b = (const float*)d_in[34];
  const float* r2w = (const float*)d_in[35];
  const float* r2b = (const float*)d_in[36];

  // workspace layout (floats)
  size_t need = ((size_t)911 * N + (size_t)4 * E) * sizeof(float);
  if (ws_size < need) return;
  float* ws      = (float*)d_ws;
  float* xl      = ws;                                   // 256N
  float* xr      = xl + (size_t)256 * N;                 // 256N
  float* agg     = xr + (size_t)256 * N;                 // 256N
  float* logits  = agg + (size_t)256 * N;                // 4(E+N)
  int*   mxo     = (int*)(logits + (size_t)4 * (E + N)); // 4N
  float* denom   = (float*)(mxo + (size_t)4 * N);        // 4N
  float* loop_ea = denom + (size_t)4 * N;                // N (also used as sm)
  float* cnt     = loop_ea + N;                          // N
  float* h0      = cnt + N;                              // 64N
  float* h1      = h0 + (size_t)64 * N;                  // 64N
  float* scores  = h1 + (size_t)64 * N;                  // N
  // buffer reuse after attention stages:
  float* gi  = xl;                        // 192N
  float* gh  = xr;                        // 192N
  float* cat = agg;                       // 128N
  float* hfb = agg + (size_t)128 * N;     // 64N
  float* hrb = agg + (size_t)192 * N;     // 64N
  float* wts   = (float*)d_out;           // N
  float* m_new = wts + N;                 // 64N

  // self-loop edge attr = mean of incoming non-self edge attrs (shared by both layers)
  fill_u32<<<cdiv(N, 256), 256, 0, stream>>>((unsigned*)cnt, 0u, N);
  fill_u32<<<cdiv(N, 256), 256, 0, stream>>>((unsigned*)loop_ea, 0u, N);
  loop_stats<<<cdiv(E, 256), 256, 0, stream>>>(src, dst, ea, cnt, loop_ea, E);
  loop_div<<<cdiv(N, 256), 256, 0, stream>>>(cnt, loop_ea, N);

  const float* xin = x;
  for (int l = 0; l < 2; l++) {
    if (l == 0) {
      gemm_launch<128>(stream, xin, Wl[0], bl[0], nullptr, nullptr, xl, N, HC, ACT_NONE);
      gemm_launch<128>(stream, xin, Wr[0], br[0], nullptr, nullptr, xr, N, HC, ACT_NONE);
    } else {
      gemm_launch<64>(stream, xin, Wl[1], bl[1], nullptr, nullptr, xl, N, HC, ACT_NONE);
      gemm_launch<64>(stream, xin, Wr[1], br[1], nullptr, nullptr, xr, N, HC, ACT_NONE);
    }

    fill_u32<<<cdiv(4 * N, 256), 256, 0, stream>>>((unsigned*)mxo, 0x80000000u, 4 * N);
    fill_u32<<<cdiv(4 * N, 256), 256, 0, stream>>>((unsigned*)denom, 0u, 4 * N);
    fill_u32<<<cdiv(256 * N, 256), 256, 0, stream>>>((unsigned*)agg, 0u, 256 * N);

    int tot4 = (E + N) * 4;
    logits_kernel<<<cdiv(tot4, 256), 256, 0, stream>>>(
        xl, xr, src, dst, ea, loop_ea, Wee[l], att[l], logits, mxo, E, N);
    denom_kernel<<<cdiv(tot4, 256), 256, 0, stream>>>(logits, dst, mxo, denom, E, N);
    scatter_kernel<<<cdiv(tot4, 256), 256, 0, stream>>>(
        xl, logits, mxo, denom, src, dst, agg, E, N);

    float* hout = (l == 0) ? h0 : h1;
    const float* resid = (l == 0) ? nullptr : h0;   // residual only when dims match
    gemm_launch<256>(stream, agg, Wp[l], bp[l], bc[l], resid, hout, N, HID, ACT_NONE);
    elu_ln<<<N, 64, 0, stream>>>(hout, g[l], be[l], N);

    xin = hout;
  }

  // GRU memory fusion
  gemm_launch<64>(stream, h1,   gru_wih, gru_bih, nullptr, nullptr, gi, N, 192, ACT_NONE);
  gemm_launch<64>(stream, pmem, gru_whh, gru_bhh, nullptr, nullptr, gh, N, 192, ACT_NONE);
  gru_kernel<<<cdiv(64 * N, 256), 256, 0, stream>>>(gi, gh, pmem, m_new, N);

  // readout
  cat_kernel<<<cdiv(128 * N, 256), 256, 0, stream>>>(h1, m_new, cat, N);
  gemm_launch<128>(stream, cat, Wf,  bfb, nullptr, nullptr, hfb, N, HID, ACT_RELU);
  gemm_launch<64>(stream,  hfb, r1w, r1b, nullptr, nullptr, hrb, N, HID, ACT_RELU);
  scores_kernel<<<cdiv(N, 256), 256, 0, stream>>>(hrb, r2w, r2b, scores, N);
  sparsemax_kernel<<<1, 1024, 0, stream>>>(scores, mask, wts, N);
}
